// MultiHeadAttention_62079457296538
// MI455X (gfx1250) — matrix operand order repaired and checked by reading, not compiled, not run on hardware
//
#include <hip/hip_runtime.h>

// ---------------------------------------------------------------------------
// MultiHeadAttention on MI455X (gfx1250): bf16 WMMA everywhere, fp32 accum.
// D=1024, H=16, kd=64, S=2048, B=2.
// Async global->LDS staging (ASYNCcnt) where the copy is un-transformed.
// ---------------------------------------------------------------------------

#define DMODEL 1024
#define SEQ    2048
#define BATCH  2
#define NHEAD  16
#define KD     64
#define MROWS  (BATCH * SEQ)   // 4096

typedef __attribute__((ext_vector_type(16))) __bf16 v16bf;
typedef __attribute__((ext_vector_type(8)))  float  v8f;

// ---------------------------------------------------------------------------
// Async global->LDS copy (gfx1250 GLOBAL_LOAD_ASYNC_TO_LDS_B128, ASYNCcnt).
// Builtin signature (from hipcc diagnostic): (int4 as1*, int4 as3*, Iimm, Iimm).
// Guarded: falls back to a plain 16-byte copy if builtins are absent.
// ---------------------------------------------------------------------------
#if __has_builtin(__builtin_amdgcn_global_load_async_to_lds_b128) && \
    __has_builtin(__builtin_amdgcn_s_wait_asynccnt)
#define HAVE_ASYNC_LDS 1
typedef int v4i_g __attribute__((vector_size(16)));
typedef __attribute__((address_space(1))) v4i_g* gv4i_p;
typedef __attribute__((address_space(3))) v4i_g* lv4i_p;
__device__ __forceinline__ void async_cp16(const void* g, void* l) {
  // low 32 bits of a generic LDS pointer == as3 offset on AMDGPU
  gv4i_p gp = (gv4i_p)(unsigned long long)(uintptr_t)g;
  lv4i_p lp = (lv4i_p)(unsigned int)(uintptr_t)l;
  __builtin_amdgcn_global_load_async_to_lds_b128(gp, lp, 0, 0);
}
__device__ __forceinline__ void async_wait() {
  __builtin_amdgcn_s_wait_asynccnt(0);
}
#else
#define HAVE_ASYNC_LDS 0
__device__ __forceinline__ void async_cp16(const void* g, void* l) {
  *(uint4*)l = *(const uint4*)g;
}
__device__ __forceinline__ void async_wait() {}
#endif

// ---------------------------------------------------------------------------
// WMMA fragment helpers (wave32; A and B share contiguous k order, not the ISA table order).
//
// A (16x32 bf16, M x K): lane(0..15) holds row M=lane, lane+16 holds same row,
//   VGPR v packs K = (lane>>4)*16 + 2v, +1.
// B (32x16 bf16, K x N): lane holds column N = lane&15, lanes 0-15 cover
//   K=0..15, lanes 16-31 cover K=16..31, VGPR v packs K = (lane>>4)*16 + 2v.
// C/D (16x16 f32): VGPR r holds row M = (lane>>4)*8 + r, col N = lane&15.
// ---------------------------------------------------------------------------

__device__ __forceinline__ v8f wmma_bf16(v16bf a, v16bf b, v8f c) {
  return __builtin_amdgcn_wmma_f32_16x16x32_bf16(
      /*neg_a=*/false, a, /*neg_b=*/false, b,
      /*c_mod=*/(short)0, c, /*reuse_a=*/false, /*reuse_b=*/false);
}

// A fragment from row-major LDS tile: elem(M,K) = base[(row0+M)*ldk + kbase + K]
__device__ __forceinline__ v16bf load_A_frag(const __bf16* base, int row0,
                                             int ldk, int kbase, int lane) {
  const int l = lane & 15, hi = lane >> 4;
  const __bf16* rp = base + (row0 + l) * ldk + kbase;
  v16bf a;
#pragma unroll
  for (int v = 0; v < 8; ++v) {
    const int kb = hi * 16 + 2 * v;
    a[2 * v]     = rp[kb];
    a[2 * v + 1] = rp[kb + 1];
  }
  return a;
}

// B fragment from row-major [K][N] LDS: elem(K,N) = base[(krow0+K)*ldn + col0+N]
__device__ __forceinline__ v16bf load_B_frag(const __bf16* base, int krow0,
                                             int col0, int ldn, int lane) {
  const int l = lane & 15, hi = lane >> 4;
  v16bf b;
#pragma unroll
  for (int v = 0; v < 8; ++v) {
    const int kr = krow0 + hi * 16 + 2 * v;
    b[2 * v]     = base[kr * ldn + col0 + l];
    b[2 * v + 1] = base[(kr + 1) * ldn + col0 + l];
  }
  return b;
}

// B fragment where the source tile is stored TRANSPOSED (row-major [N][K]):
// elem(K,N) = base[(nrow0+N)*ldk + kbase + K]  -> 16 contiguous bf16 per lane.
__device__ __forceinline__ v16bf load_BT_frag(const __bf16* base, int nrow0,
                                              int ldk, int kbase, int lane) {
  const int l = lane & 15, hi = lane >> 4;
  const __bf16* rp = base + (nrow0 + l) * ldk + kbase + hi * 16;
  v16bf b;
#pragma unroll
  for (int v = 0; v < 16; ++v) b[v] = rp[v];
  return b;
}

// ---------------------------------------------------------------------------
// Kernel 1: convert the 4 fp32 weight matrices [N][K] into bf16 W^T [K][N].
// ---------------------------------------------------------------------------
__global__ void __launch_bounds__(256)
wtrans_kernel(const float* __restrict__ W0, const float* __restrict__ W1,
              const float* __restrict__ W2, const float* __restrict__ W3,
              __bf16* __restrict__ T) {
  const int idx   = blockIdx.x * 256 + threadIdx.x;  // n*D + k  (coalesced read)
  const int which = blockIdx.y;
  const float* W  = (which == 0) ? W0 : (which == 1) ? W1 : (which == 2) ? W2 : W3;
  const int n = idx >> 10;          // D = 1024
  const int k = idx & (DMODEL - 1);
  T[(size_t)which * DMODEL * DMODEL + (size_t)k * DMODEL + n] = (__bf16)W[idx];
}

// ---------------------------------------------------------------------------
// Kernel 2/4: GEMM  P = X @ W^T + bias   (M=4096, N=K=1024), bf16 WMMA.
// MODE 0: X fp32 (convert in staging), output bf16 head-major [b,h,s,kd].
// MODE 1: X bf16 (async-staged),       output fp32 plain [M][N].
// Block: 256 thr (8 waves), tile 128(M) x 64(N); wave tile 32x32; K-step 32.
// ---------------------------------------------------------------------------
template <int MODE>
__global__ void __launch_bounds__(256)
gemm_kernel(const void* __restrict__ Xin, const __bf16* __restrict__ Wt,
            const float* __restrict__ bias, void* __restrict__ Out) {
  __shared__ __bf16 Xs[128 * 48];  // 128 x 32, pitch 48 (96B rows, 16B aligned)
  __shared__ __bf16 Ws[32 * 72];   // 32 x 64, pitch 72 (144B rows, 16B aligned)

  const int tid  = threadIdx.x;
  const int lane = tid & 31;
  const int w    = tid >> 5;
  const int wm   = w & 3;   // 4 waves along M
  const int wn   = w >> 2;  // 2 waves along N
  const int m0   = blockIdx.x * 128;
  const int n0   = blockIdx.y * 64;

  v8f acc[2][2] = {};

  for (int kk = 0; kk < DMODEL; kk += 32) {
    __syncthreads();
    // ---- stage X tile 128x32 ----
    if (MODE == 0) {
      const float* X = (const float*)Xin;
#pragma unroll
      for (int i = 0; i < 16; ++i) {
        const int e = tid + i * 256;       // 0..4095
        const int r = e >> 5, c = e & 31;
        Xs[r * 48 + c] = (__bf16)X[(size_t)(m0 + r) * DMODEL + kk + c];
        if ((i & 3) == 0 && kk + 32 < DMODEL)   // warm L2 for next K-step
          __builtin_prefetch(&X[(size_t)(m0 + r) * DMODEL + kk + 32 + c], 0, 1);
      }
    } else {
      const __bf16* X = (const __bf16*)Xin;
#pragma unroll
      for (int i = 0; i < 2; ++i) {
        const int e = tid + i * 256;       // 0..511  (8-elem chunks)
        const int r = e >> 2, c = (e & 3) * 8;
        async_cp16(&X[(size_t)(m0 + r) * DMODEL + kk + c], &Xs[r * 48 + c]);
      }
    }
    // ---- stage W^T tile 32x64 (pure bf16 copy -> async DMA) ----
    {
      const int r = tid >> 3, c = (tid & 7) * 8;   // 256 chunks of 16B
      async_cp16(&Wt[(size_t)(kk + r) * DMODEL + n0 + c], &Ws[r * 72 + c]);
    }
    async_wait();
    __syncthreads();

    v16bf bfr[2];
#pragma unroll
    for (int ni = 0; ni < 2; ++ni)
      bfr[ni] = load_B_frag(Ws, 0, wn * 32 + ni * 16, 72, lane);
#pragma unroll
    for (int mi = 0; mi < 2; ++mi) {
      v16bf a = load_A_frag(Xs, wm * 32 + mi * 16, 48, 0, lane);
#pragma unroll
      for (int ni = 0; ni < 2; ++ni)
        acc[mi][ni] = wmma_bf16(a, bfr[ni], acc[mi][ni]);
    }
  }

  // ---- epilogue: bias + store ----
  const int l = lane & 15, hi = lane >> 4;
#pragma unroll
  for (int mi = 0; mi < 2; ++mi) {
#pragma unroll
    for (int ni = 0; ni < 2; ++ni) {
      const int col = n0 + wn * 32 + ni * 16 + l;
      const float bv = bias[col];
#pragma unroll
      for (int r = 0; r < 8; ++r) {
        const int row = m0 + wm * 32 + mi * 16 + hi * 8 + r;
        const float v = acc[mi][ni][r] + bv;
        if (MODE == 0) {
          const int b = row >> 11;            // S = 2048
          const int s = row & (SEQ - 1);
          const int h = col >> 6;             // kd = 64
          const int d = col & (KD - 1);
          ((__bf16*)Out)[(((size_t)(b * NHEAD + h) * SEQ + s) * KD) + d] =
              (__bf16)v;
        } else {
          ((float*)Out)[(size_t)row * DMODEL + col] = v;
        }
      }
    }
  }
}

// ---------------------------------------------------------------------------
// Kernel 3: flash attention per (b,h).  Block = 128 thr (4 waves), 64 q-rows.
// Each wave owns 16 q-rows x full kd=64, iterates 64-key blocks with online
// softmax.  QK^T and PV both via bf16 WMMA.  Q/K staged via async DMA.
// ---------------------------------------------------------------------------
__global__ void __launch_bounds__(128)
attn_kernel(const __bf16* __restrict__ Qh, const __bf16* __restrict__ Kh,
            const __bf16* __restrict__ Vh, __bf16* __restrict__ AO) {
  __shared__ __bf16 Qs[64 * 72];       // Q block [qrow][d]
  __shared__ __bf16 Ks[64 * 72];       // K block [krow][d]
  __shared__ __bf16 Vt[64 * 72];       // V block transposed [d][krow]
  __shared__ __bf16 Ps[4 * 16 * 72];   // per-wave P tile [16][64]

  const int tid  = threadIdx.x;
  const int lane = tid & 31;
  const int w    = tid >> 5;
  const int l    = lane & 15, hi = lane >> 4;
  const int q0   = blockIdx.x * 64;
  const int bh   = blockIdx.y;                  // b*H + h
  const size_t base = (size_t)bh * SEQ * KD;

  // ---- stage Q block once (64x64 bf16) via async DMA ----
#pragma unroll
  for (int i = 0; i < 4; ++i) {
    const int e = tid + i * 128;       // 0..511 (8-elem chunks)
    const int r = e >> 3, c = (e & 7) * 8;
    async_cp16(&Qh[base + (size_t)(q0 + r) * KD + c], &Qs[r * 72 + c]);
  }

  v8f O[4] = {};
  float rmax[8], rsum[8];
#pragma unroll
  for (int r = 0; r < 8; ++r) { rmax[r] = -1e30f; rsum[r] = 0.f; }

  __bf16* Pw = Ps + w * (16 * 72);

  for (int kb0 = 0; kb0 < SEQ; kb0 += 64) {
    __syncthreads();
    // ---- stage K (async, row-major) and V (manual transpose) ----
#pragma unroll
    for (int i = 0; i < 4; ++i) {
      const int e = tid + i * 128;
      const int r = e >> 3, c = (e & 7) * 8;
      async_cp16(&Kh[base + (size_t)(kb0 + r) * KD + c], &Ks[r * 72 + c]);
      if (kb0 + 64 < SEQ) {           // warm L2 for the next key block
        __builtin_prefetch(&Kh[base + (size_t)(kb0 + 64 + r) * KD + c], 0, 1);
        __builtin_prefetch(&Vh[base + (size_t)(kb0 + 64 + r) * KD + c], 0, 1);
      }
    }
#pragma unroll
    for (int i = 0; i < 32; ++i) {
      const int e = tid + i * 128;
      const int r = e >> 6, c = e & 63;
      Vt[c * 72 + r] = Vh[base + (size_t)(kb0 + r) * KD + c];
    }
    async_wait();
    __syncthreads();

    // ---- S = Q K^T : 4 tiles of 16 key-cols, K-dim = kd = 64 ----
    v8f Sc[4] = {};
#pragma unroll
    for (int kk = 0; kk < KD; kk += 32) {
      v16bf a = load_A_frag(Qs, w * 16, 72, kk, lane);
#pragma unroll
      for (int ni = 0; ni < 4; ++ni) {
        v16bf b = load_BT_frag(Ks, ni * 16, 72, kk, lane);  // B[d][kc]=K[kc][d]
        Sc[ni] = wmma_bf16(a, b, Sc[ni]);
      }
    }

    // ---- online softmax (scale = 1/sqrt(64) = 0.125) ----
#pragma unroll
    for (int r = 0; r < 8; ++r) {
      float cur = -1e30f;
#pragma unroll
      for (int ni = 0; ni < 4; ++ni) cur = fmaxf(cur, Sc[ni][r] * 0.125f);
#pragma unroll
      for (int m = 1; m <= 8; m <<= 1) cur = fmaxf(cur, __shfl_xor(cur, m, 32));
      const float mnew  = fmaxf(rmax[r], cur);
      const float alpha = __expf(rmax[r] - mnew);
      rmax[r] = mnew;
      float ps = 0.f;
#pragma unroll
      for (int ni = 0; ni < 4; ++ni) {
        const float p = __expf(Sc[ni][r] * 0.125f - mnew);
        Sc[ni][r] = p;
        ps += p;
      }
#pragma unroll
      for (int m = 1; m <= 8; m <<= 1) ps += __shfl_xor(ps, m, 32);
      rsum[r] = rsum[r] * alpha + ps;
#pragma unroll
      for (int ni = 0; ni < 4; ++ni) O[ni][r] *= alpha;
    }

    // ---- stash P as bf16 A-operand tile (wave-private LDS) ----
#pragma unroll
    for (int ni = 0; ni < 4; ++ni)
#pragma unroll
      for (int r = 0; r < 8; ++r)
        Pw[(hi * 8 + r) * 72 + ni * 16 + l] = (__bf16)Sc[ni][r];

    // ---- O += P V : A = P[16x64], B = V[64x64] (staged transposed) ----
#pragma unroll
    for (int kk = 0; kk < 64; kk += 32) {
      v16bf a = load_A_frag(Pw, 0, 72, kk, lane);
#pragma unroll
      for (int ni = 0; ni < 4; ++ni) {
        v16bf b = load_BT_frag(Vt, ni * 16, 72, kk, lane);  // B[kr][dc]=V[kr][dc]
        O[ni] = wmma_bf16(a, b, O[ni]);
      }
    }
  }

  // ---- normalize & write to [b, s, h*64+d] (bf16, feeds output proj) ----
  const int b = bh >> 4;          // H = 16
  const int h = bh & (NHEAD - 1);
#pragma unroll
  for (int ni = 0; ni < 4; ++ni) {
    const int dcol = ni * 16 + l;
#pragma unroll
    for (int r = 0; r < 8; ++r) {
      const int qrow = q0 + w * 16 + hi * 8 + r;
      const float v  = O[ni][r] / rsum[r];
      AO[((size_t)(b * SEQ + qrow)) * DMODEL + h * KD + dcol] = (__bf16)v;
    }
  }
}

// ---------------------------------------------------------------------------
// Host-side launcher
// ---------------------------------------------------------------------------
extern "C" void kernel_launch(void* const* d_in, const int* in_sizes, int n_in,
                              void* d_out, int out_size, void* d_ws,
                              size_t ws_size, hipStream_t stream) {
  (void)in_sizes; (void)n_in; (void)out_size; (void)ws_size;

  const float* query = (const float*)d_in[0];
  const float* value = (const float*)d_in[1];
  const float* key   = (const float*)d_in[2];
  const float* Wq_w  = (const float*)d_in[3];
  const float* Wq_b  = (const float*)d_in[4];
  const float* Wk_w  = (const float*)d_in[5];
  const float* Wk_b  = (const float*)d_in[6];
  const float* Wv_w  = (const float*)d_in[7];
  const float* Wv_b  = (const float*)d_in[8];
  const float* Wo_w  = (const float*)d_in[9];
  const float* Wo_b  = (const float*)d_in[10];

  const size_t DD = (size_t)DMODEL * DMODEL;       // 1M elems
  const size_t MD = (size_t)MROWS * DMODEL;        // 4M elems

  __bf16* WT = (__bf16*)d_ws;        // 4 transposed weights, bf16
  __bf16* Qh = WT + 4 * DD;          // head-major Q  [b,h,s,kd]
  __bf16* Kh = Qh + MD;
  __bf16* Vh = Kh + MD;
  __bf16* AO = Vh + MD;              // attention out [b,s,h*kd]

  // 1) weights -> bf16 transposed
  wtrans_kernel<<<dim3(DD / 256, 4), 256, 0, stream>>>(Wq_w, Wk_w, Wv_w, Wo_w, WT);

  // 2) Q/K/V projections (fp32 in -> bf16 head-major out)
  const dim3 gg(MROWS / 128, DMODEL / 64);
  gemm_kernel<0><<<gg, 256, 0, stream>>>(query, WT + 0 * DD, Wq_b, Qh);
  gemm_kernel<0><<<gg, 256, 0, stream>>>(key,   WT + 1 * DD, Wk_b, Kh);
  gemm_kernel<0><<<gg, 256, 0, stream>>>(value, WT + 2 * DD, Wv_b, Vh);

  // 3) flash attention per (b,h)
  attn_kernel<<<dim3(SEQ / 64, BATCH * NHEAD), 128, 0, stream>>>(Qh, Kh, Vh, AO);

  // 4) output projection (bf16 in -> fp32 out)
  gemm_kernel<1><<<gg, 256, 0, stream>>>(AO, WT + 3 * DD, Wo_b, (float*)d_out);
}
